// MCTS_26242250178672
// MI455X (gfx1250) — compile-verified, matrix operand order checked by reading, not run on hardware
//
#include <hip/hip_runtime.h>
#include <math.h>

// ---------------------------------------------------------------------------
// CDNA5 (gfx1250) bf16 WMMA attention pipeline.
// All GEMMs use v_wmma_f32_16x16x32_bf16 (wave32, 16x16 tiles, K=32).
// Round 4: double-buffered LDS pipeline (1 barrier per K-step), 128-bit
// LDS staging & fragment loads (LDSP=40 keeps 16B alignment), padded ld=200
// for the two N=196-pitch tensors so global b128 loads stay aligned.
// ---------------------------------------------------------------------------

typedef __attribute__((ext_vector_type(16))) __bf16 v16bf;
typedef __attribute__((ext_vector_type(8)))  float  v8f;

#define BM 64
#define BN 64
#define BKK 32
#define LDSP 40   // LDS row pitch in bf16 (80B): keeps every half*8 offset 16B-aligned

// --------------------------- elementwise helpers ---------------------------

__global__ void cvt_f32_bf16(const float* __restrict__ s, __bf16* __restrict__ d, long n) {
    long i = (long)blockIdx.x * blockDim.x + threadIdx.x;
    if (i < n) d[i] = (__bf16)s[i];
}

__global__ void zero_f32(float* __restrict__ p, long n) {
    long i = (long)blockIdx.x * blockDim.x + threadIdx.x;
    if (i < n) p[i] = 0.0f;
}

// --------------------------- generic WMMA GEMM ------------------------------
// D[M x N] (+)= scale * A[M x K] * B[N x K]^T     (A,B bf16 row-major)
// batch index z = blockIdx.z;  b = z / Hdiv, h = z % Hdiv
// TRANS : store D[n*ldd + m] instead of D[m*ldd + n]
// ATOM  : fp32 atomicAdd accumulation (fuses the mean-over-heads)
// OBF16 : store bf16 instead of fp32
//
// Requirements (all call sites satisfy them):
//  - lda/ldb multiples of 8 (16B-aligned rows), A/B 16B-aligned ws pointers
//  - Kd even; K tail (Kd%32) handled by a peeled zero-filled iteration
//  - out-of-range rows are CLAMPED, not masked: they only affect D rows/cols
//    that the edge store never writes
//  - loader may read up to 62B past a region's last row: workspace is padded

template<int TRANS, int ATOM, int OBF16>
__global__ void __launch_bounds__(128)
gemm_abt(const __bf16* __restrict__ A, const __bf16* __restrict__ B,
         void* __restrict__ Dp,
         int M, int N, int Kd,
         int lda, int ldb, int ldd,
         long asb, long ash, long bsb, long bsh, long dsb, long dsh,
         int Hdiv, float scale)
{
    __shared__ __bf16 sA[2 * BM * LDSP];   // ping-pong buffers
    __shared__ __bf16 sB[2 * BN * LDSP];

    const int tid  = threadIdx.x;
    const int wave = tid >> 5;
    const int lane = tid & 31;
    const int z = blockIdx.z;
    const int b = z / Hdiv, h = z % Hdiv;

    A += (long)b * asb + (long)h * ash;
    B += (long)b * bsb + (long)h * bsh;
    float*  Df = (float*)Dp  + (long)b * dsb + (long)h * dsh;
    __bf16* Db = (__bf16*)Dp + (long)b * dsb + (long)h * dsh;

    const int m0 = blockIdx.y * BM;
    const int n0 = blockIdx.x * BN;

    const int wm = (wave >> 1) * 32;   // wave's 32x32 sub-tile within 64x64
    const int wn = (wave & 1) * 32;

    v8f acc[2][2] = {};

    // cooperative loader mapping: 128 threads, each covers one row, 16 elems
    const int lrow = tid >> 1;
    const int lcol = (tid & 1) * 16;

    // clamped (always in-bounds) row pointers; no masking needed (see header)
    const int gm = m0 + lrow;
    const int gn = n0 + lrow;
    const int gmc = (gm < M) ? gm : (M - 1);
    const int gnc = (gn < N) ? gn : (N - 1);
    const __bf16* Arow = A + (long)gmc * lda + lcol;   // 16B aligned
    const __bf16* Brow = B + (long)gnc * ldb + lcol;   // 16B aligned

    // staged tile registers (tile t+1 while computing tile t)
    uint4 ra0, ra1, rb0, rb1;
    auto gload = [&](int k0) {
        const __bf16* ap = Arow + k0;
        const __bf16* bp = Brow + k0;
        ra0 = *(const uint4*)(ap);
        ra1 = *(const uint4*)(ap + 8);
        rb0 = *(const uint4*)(bp);
        rb1 = *(const uint4*)(bp + 8);
        if (k0 + BKK < Kd) {                    // look one more tile ahead
            __builtin_prefetch(ap + BKK, 0, 1);
            __builtin_prefetch(bp + BKK, 0, 1);
        }
    };
    auto lstore = [&](int buf) {
        __bf16* a = sA + buf * (BM * LDSP) + lrow * LDSP + lcol;
        __bf16* bq = sB + buf * (BN * LDSP) + lrow * LDSP + lcol;
        *(uint4*)(a)     = ra0;
        *(uint4*)(a + 8) = ra1;
        *(uint4*)(bq)     = rb0;
        *(uint4*)(bq + 8) = rb1;
    };

    // ---- fragment-build (2x ds_load_b128 per fragment) + 4x WMMA ----
    const int half = lane >> 4;   // K-octet select
    const int l15  = lane & 15;   // M (A) / N (B) index
    auto frag_mma = [&](int buf) {
        const __bf16* ab = sA + buf * (BM * LDSP);
        const __bf16* bb = sB + buf * (BN * LDSP);
        union Fr { v16bf v; uint4 q[2]; };
        Fr fa[2], fb[2];
#pragma unroll
        for (int i = 0; i < 2; ++i) {
            const __bf16* base = ab + (wm + i * 16 + l15) * LDSP + half * 8;
            fa[i].q[0] = *(const uint4*)(base);
            fa[i].q[1] = *(const uint4*)(base + 16);
        }
#pragma unroll
        for (int j = 0; j < 2; ++j) {
            const __bf16* base = bb + (wn + j * 16 + l15) * LDSP + half * 8;
            fb[j].q[0] = *(const uint4*)(base);
            fb[j].q[1] = *(const uint4*)(base + 16);
        }
#pragma unroll
        for (int i = 0; i < 2; ++i)
#pragma unroll
            for (int j = 0; j < 2; ++j)
                acc[i][j] = __builtin_amdgcn_wmma_f32_16x16x32_bf16(
                    false, fa[i].v, false, fb[j].v,
                    (short)0, acc[i][j], false, false);
    };

    const int KdMain = Kd & ~(BKK - 1);
    const int nT = KdMain / BKK;

    // ---- double-buffered main pipeline: one barrier per K-step ----
    if (nT > 0) {
        gload(0);
        lstore(0);
        if (nT > 1) gload(BKK);
        __syncthreads();
#pragma unroll 2
        for (int t = 0; t < nT; ++t) {
            frag_mma(t & 1);                       // ds_load + wmma on buf t
            if (t + 1 < nT) lstore((t + 1) & 1);   // stage tile t+1 (other buf)
            if (t + 2 < nT) gload((t + 2) * BKK);  // fetch tile t+2 into regs
            __syncthreads();
        }
    }

    // ---- K tail (Kd % 32 != 0, e.g. Kd = 196): zero-fill invalid pairs ----
    if (KdMain < Kd) {
        const int buf = nT & 1;
        unsigned int* aw = (unsigned int*)(sA + buf * (BM * LDSP) + lrow * LDSP + lcol);
        unsigned int* bw = (unsigned int*)(sB + buf * (BN * LDSP) + lrow * LDSP + lcol);
        const __bf16* ap = Arow + KdMain;
        const __bf16* bp = Brow + KdMain;
#pragma unroll
        for (int p = 0; p < 8; ++p) {
            const bool kok = (KdMain + lcol + p * 2 + 1) < Kd;
            const unsigned int va = *(const unsigned int*)(ap + p * 2);
            const unsigned int vb = *(const unsigned int*)(bp + p * 2);
            aw[p] = kok ? va : 0u;
            bw[p] = kok ? vb : 0u;
        }
        __syncthreads();
        frag_mma(buf);
    }

    // ---- store D per 16x16 f32 C/D VGPR layout ----
    const bool fullTile = (m0 + BM <= M) && (n0 + BN <= N);

    if (fullTile) {
        // interior tile: straight-line stores, no per-lane predication
#pragma unroll
        for (int i = 0; i < 2; ++i) {
#pragma unroll
            for (int j = 0; j < 2; ++j) {
                const int sm = m0 + wm + i * 16 + half * 8;
                const int sn = n0 + wn + j * 16 + l15;
#pragma unroll
                for (int r = 0; r < 8; ++r) {
                    const float val = acc[i][j][r] * scale;
                    const long idx = TRANS ? ((long)sn * ldd + (sm + r))
                                           : ((long)(sm + r) * ldd + sn);
                    if (ATOM)       atomicAdd(Df + idx, val);
                    else if (OBF16) Db[idx] = (__bf16)val;
                    else            Df[idx] = val;
                }
            }
        }
    } else {
#pragma unroll
        for (int i = 0; i < 2; ++i) {
#pragma unroll
            for (int j = 0; j < 2; ++j) {
#pragma unroll
                for (int r = 0; r < 8; ++r) {
                    const int gmm = m0 + wm + i * 16 + half * 8 + r;
                    const int gnn = n0 + wn + j * 16 + l15;
                    if (gmm < M && gnn < N) {
                        const float val = acc[i][j][r] * scale;
                        const long idx = TRANS ? ((long)gnn * ldd + gmm)
                                               : ((long)gmm * ldd + gnn);
                        if (ATOM)       atomicAdd(Df + idx, val);
                        else if (OBF16) Db[idx] = (__bf16)val;
                        else            Df[idx] = val;
                    }
                }
            }
        }
    }
}

// ------------------- InstanceNorm stats (per b,h over C*KV) -----------------

__global__ void __launch_bounds__(256)
bh_stats(const float* __restrict__ s, float* __restrict__ stats,
         long per, float inv_n)
{
    __shared__ float red0[256];
    __shared__ float red1[256];
    const long base = (long)blockIdx.x * per;
    float sum = 0.f, sq = 0.f;
    for (long i = threadIdx.x; i < per; i += 256) {
        const float x = s[base + i];
        sum += x; sq += x * x;
    }
    red0[threadIdx.x] = sum; red1[threadIdx.x] = sq;
    __syncthreads();
    for (int o = 128; o > 0; o >>= 1) {
        if ((int)threadIdx.x < o) {
            red0[threadIdx.x] += red0[threadIdx.x + o];
            red1[threadIdx.x] += red1[threadIdx.x + o];
        }
        __syncthreads();
    }
    if (threadIdx.x == 0) {
        const float m = red0[0] * inv_n;
        const float v = red1[0] * inv_n - m * m;
        stats[blockIdx.x * 2 + 0] = m;
        stats[blockIdx.x * 2 + 1] = rsqrtf(v + 1e-5f);
    }
}

// ------------- normalize + row softmax, write bf16 probabilities ------------

__global__ void __launch_bounds__(256)
inorm_softmax(const float* __restrict__ s, const float* __restrict__ stats,
              __bf16* __restrict__ p, int C, int KVd)
{
    const int row = blockIdx.x;          // row = z*C + c,  z = b*H + h
    const int z = row / C;
    const float m    = stats[z * 2 + 0];
    const float rstd = stats[z * 2 + 1];
    const float* sr = s + (long)row * KVd;
    __bf16*      pr = p + (long)row * KVd;
    const int t = threadIdx.x;

    float x[4];
    float mx = -3.4e38f;
#pragma unroll
    for (int j = 0; j < 4; ++j) {
        const int i = t + j * 256;
        x[j] = (i < KVd) ? (sr[i] - m) * rstd : -3.4e38f;
        mx = fmaxf(mx, x[j]);
    }
    __shared__ float red[256];
    red[t] = mx; __syncthreads();
    for (int o = 128; o > 0; o >>= 1) {
        if (t < o) red[t] = fmaxf(red[t], red[t + o]);
        __syncthreads();
    }
    mx = red[0]; __syncthreads();

    float sum = 0.f;
#pragma unroll
    for (int j = 0; j < 4; ++j) {
        const int i = t + j * 256;
        if (i < KVd) { x[j] = __expf(x[j] - mx); sum += x[j]; }
    }
    red[t] = sum; __syncthreads();
    for (int o = 128; o > 0; o >>= 1) {
        if (t < o) red[t] += red[t + o];
        __syncthreads();
    }
    const float inv = 1.0f / red[0];
#pragma unroll
    for (int j = 0; j < 4; ++j) {
        const int i = t + j * 256;
        if (i < KVd) pr[i] = (__bf16)(x[j] * inv);
    }
}

// ------------------------------- host driver --------------------------------

extern "C" void kernel_launch(void* const* d_in, const int* in_sizes, int n_in,
                              void* d_out, int out_size, void* d_ws, size_t ws_size,
                              hipStream_t stream)
{
    (void)in_sizes; (void)n_in; (void)out_size; (void)ws_size;

    const int Bb = 32, Nn = 196, Hh = 4, KVd = 960;
    const int NnP = 200;                  // padded pitch for N=196 tensors (%8==0)
    const int Cs[4] = {64, 128, 256, 512};

    const float* emb[4];  for (int i = 0; i < 4; ++i) emb[i] = (const float*)d_in[i];
    const float* embAll = (const float*)d_in[4];
    const float* Wq[4];   for (int i = 0; i < 4; ++i) Wq[i]  = (const float*)d_in[5 + i];
    const float* Wk = (const float*)d_in[9];
    const float* Wv = (const float*)d_in[10];
    const float* Wo[4];   for (int i = 0; i < 4; ++i) Wo[i]  = (const float*)d_in[11 + i];
    float* out = (float*)d_out;

    // ---- carve workspace (256B-aligned regions + tail pad for overshoot) ----
    char* ws = (char*)d_ws;
    size_t off = 0;
    auto alloc = [&](size_t bytes) -> char* {
        char* p = ws + off;
        off = (off + bytes + 255) & ~(size_t)255;
        return p;
    };
    __bf16* embAll_b = (__bf16*)alloc((size_t)Bb * Nn * KVd * 2);
    __bf16* emb_b[4];
    for (int i = 0; i < 4; ++i) emb_b[i] = (__bf16*)alloc((size_t)Bb * Nn * Cs[i] * 2);
    __bf16* Wq_b[4];
    for (int i = 0; i < 4; ++i) Wq_b[i] = (__bf16*)alloc((size_t)Hh * Cs[i] * Cs[i] * 2);
    __bf16* Wk_b = (__bf16*)alloc((size_t)Hh * KVd * KVd * 2);
    __bf16* Wv_b = (__bf16*)alloc((size_t)Hh * KVd * KVd * 2);
    __bf16* Wo_b[4];
    for (int i = 0; i < 4; ++i) Wo_b[i] = (__bf16*)alloc((size_t)Cs[i] * Cs[i] * 2);
    __bf16* Kt  = (__bf16*)alloc((size_t)Bb * Hh * KVd * NnP * 2);  // [b,h][o][n] ld=200
    __bf16* Vm  = (__bf16*)alloc((size_t)Bb * Hh * Nn * KVd * 2);   // [b,h][n][o]
    __bf16* Qt  = (__bf16*)alloc((size_t)Bb * Hh * 512 * NnP * 2);  // [b,h][c][n] ld=200
    float*  S   = (float*)alloc((size_t)Bb * Hh * 512 * KVd * 4);   // [b,h][c][o]
    __bf16* P   = (__bf16*)alloc((size_t)Bb * Hh * 512 * KVd * 2);  // probs bf16
    float*  stats = (float*)alloc((size_t)Bb * Hh * 2 * 4);
    float*  ctx   = (float*)alloc((size_t)Bb * Nn * 512 * 4);       // [b][n][c]
    __bf16* ctxb  = (__bf16*)alloc((size_t)Bb * Nn * 512 * 2);
    (void)alloc(256);  // tail pad for K-overshoot reads

    auto cvt = [&](const float* s_, __bf16* d_, long n) {
        cvt_f32_bf16<<<(unsigned)((n + 255) / 256), 256, 0, stream>>>(s_, d_, n);
    };

    // ---- fp32 -> bf16 for everything feeding WMMA ----
    cvt(embAll, embAll_b, (long)Bb * Nn * KVd);
    for (int i = 0; i < 4; ++i) {
        cvt(emb[i], emb_b[i], (long)Bb * Nn * Cs[i]);
        cvt(Wq[i],  Wq_b[i],  (long)Hh * Cs[i] * Cs[i]);
        cvt(Wo[i],  Wo_b[i],  (long)Cs[i] * Cs[i]);
    }
    cvt(Wk, Wk_b, (long)Hh * KVd * KVd);
    cvt(Wv, Wv_b, (long)Hh * KVd * KVd);

    // ---- K projection, stored transposed [b,h][o][n]; V kept [b,h][n][o] ----
    {
        dim3 g((KVd + 63) / 64, (Nn + 63) / 64, Bb * Hh);
        gemm_abt<1, 0, 1><<<g, 128, 0, stream>>>(embAll_b, Wk_b, Kt,
            Nn, KVd, KVd, KVd, KVd, NnP,
            (long)Nn * KVd, 0, 0, (long)KVd * KVd,
            (long)Hh * KVd * NnP, (long)KVd * NnP, Hh, 1.0f);
        gemm_abt<0, 0, 1><<<g, 128, 0, stream>>>(embAll_b, Wv_b, Vm,
            Nn, KVd, KVd, KVd, KVd, KVd,
            (long)Nn * KVd, 0, 0, (long)KVd * KVd,
            (long)Hh * Nn * KVd, (long)Nn * KVd, Hh, 1.0f);
    }

    const float sscale = 1.0f / sqrtf((float)KVd);
    long outoff = 0;

    for (int br = 0; br < 4; ++br) {
        const int C = Cs[br];

        // Qt[b,h] (C x N) = Wq[h] (C x C) @ emb[b]^T   (ld = 200)
        dim3 g1((Nn + 63) / 64, (C + 63) / 64, Bb * Hh);
        gemm_abt<0, 0, 1><<<g1, 128, 0, stream>>>(Wq_b[br], emb_b[br], Qt,
            C, Nn, C, C, C, NnP,
            0, (long)C * C, (long)Nn * C, 0,
            (long)Hh * C * NnP, (long)C * NnP, Hh, 1.0f);

        // S[b,h] (C x KV) = Qt (C x N) @ Kt (KV x N)^T, scaled 1/sqrt(KV)
        dim3 g2((KVd + 63) / 64, (C + 63) / 64, Bb * Hh);
        gemm_abt<0, 0, 0><<<g2, 128, 0, stream>>>(Qt, Kt, S,
            C, KVd, Nn, NnP, NnP, KVd,
            (long)Hh * C * NnP, (long)C * NnP,
            (long)Hh * KVd * NnP, (long)KVd * NnP,
            (long)Hh * C * KVd, (long)C * KVd, Hh, sscale);

        // InstanceNorm stats per (b,h)
        bh_stats<<<Bb * Hh, 256, 0, stream>>>(S, stats, (long)C * KVd,
                                              1.0f / ((float)C * (float)KVd));

        // zero the ctx accumulator (atomic target)
        const long nctx = (long)Bb * Nn * C;
        zero_f32<<<(unsigned)((nctx + 255) / 256), 256, 0, stream>>>(ctx, nctx);

        // normalize + softmax over KV, emit bf16 probs
        inorm_softmax<<<Bb * Hh * C, 256, 0, stream>>>(S, stats, P, C, KVd);

        // ctx[b][n][c] += (1/H) * (P (C x KV) @ V (N x KV)^T)^T  — atomic over h
        dim3 g3((Nn + 63) / 64, (C + 63) / 64, Bb * Hh);
        gemm_abt<1, 1, 0><<<g3, 128, 0, stream>>>(P, Vm, ctx,
            C, Nn, KVd, KVd, KVd, C,
            (long)Hh * C * KVd, (long)C * KVd,
            (long)Hh * Nn * KVd, (long)Nn * KVd,
            (long)Nn * C, 0, Hh, 0.25f);

        // ctx -> bf16
        cvt(ctx, ctxb, nctx);

        // O[b] (N x C) = ctx[b] (N x C) @ Wo (C x C)^T  -> fp32 into d_out
        dim3 g4((C + 63) / 64, (Nn + 63) / 64, Bb);
        gemm_abt<0, 0, 0><<<g4, 128, 0, stream>>>(ctxb, Wo_b[br], out + outoff,
            Nn, C, C, C, C, C,
            (long)Nn * C, 0, 0, 0,
            (long)Nn * C, 0, 1, 1.0f);

        outoff += (long)Bb * Nn * C;
    }
}
